// SigMMD_naive_2611340116458
// MI455X (gfx1250) — matrix operand order repaired and checked by reading, not compiled, not run on hardware
//
#include <hip/hip_runtime.h>

typedef __attribute__((ext_vector_type(16))) _Float16 v16h;
typedef __attribute__((ext_vector_type(8)))  _Float16 v8h;
typedef __attribute__((ext_vector_type(8)))  float    v8f;

#define N_PTS  128
#define DIM    16
#define KPAD   32    // K padded to WMMA depth; cols 16..31 are zeros
#define MMDIM  127
#define STRIDE 130   // stride-1 = 129 odd -> anti-diagonal LDS reads walk banks

__global__ __launch_bounds__(128)
void sig_pde_fused_kernel(const float* __restrict__ X,
                          const float* __restrict__ Y,
                          float* __restrict__ Kout) {
  __shared__ __align__(64) float    inc[N_PTS * STRIDE];     // 66560 B (also staging)
  __shared__ __align__(64) _Float16 dXh[N_PTS][KPAD];        // 8192 B, zero-padded K
  __shared__ __align__(64) _Float16 dYh[N_PTS][KPAD];        // 8192 B, zero-padded K
  __shared__ __align__(16) float    diag[3][N_PTS + 1];      // +1: leading zero elem

  const int tid  = threadIdx.x;
  const int blk  = blockIdx.x;
  const int type = blk >> 10;        // 0: XX, 1: YY, 2: XY
  const int pair = blk & 1023;
  const int a = pair >> 5, b = pair & 31;

  const float* Xp = (type == 1) ? Y : X;
  const float* Yp = (type == 0) ? X : Y;
  const float* xa = Xp + a * (N_PTS * DIM);
  const float* yb = Yp + b * (N_PTS * DIM);

  // ---- Phase 1: stage paths (inc[] as scratch), form zero-padded f16 increments ----
  float* xs = inc;                 // 2048 floats
  float* ys = inc + N_PTS * DIM;   // 2048 floats
  for (int i = tid; i < N_PTS * DIM; i += 128) { xs[i] = xa[i]; ys[i] = yb[i]; }
  __syncthreads();
  for (int i = tid; i < N_PTS * KPAD; i += 128) {
    int m = i >> 5, k = i & 31;
    bool live = (k < DIM) && (m < MMDIM);
    int base = m * DIM + (live ? k : 0);
    float dx = live ? (xs[base + DIM] - xs[base]) : 0.0f;
    float dy = live ? (ys[base + DIM] - ys[base]) : 0.0f;
    dXh[m][k] = (_Float16)dx;
    dYh[m][k] = (_Float16)dy;
  }
  __syncthreads();

  // ---- Phase 2: inc = dX * dY^T via WMMA; each wave: 2 tile-rows x 8 tile-cols ----
  const int wave = tid >> 5;
  const int lane = tid & 31;
  const int lrow = lane & 15;
  const int half = lane >> 4;
  for (int trs = 0; trs < 2; ++trs) {
    const int mb = (wave * 2 + trs) * 16;
    // A fragment (16x32 f16): lane holds row mb+lrow, K = half*16 + (0..15)
    union { v16h v; v8h h[2]; } au;
    const _Float16* arow = &dXh[mb + lrow][0];
    au.h[0] = *(const v8h*)(arow + half * 16);
    au.h[1] = *(const v8h*)(arow + half * 16 + 8);
    const v16h afrag = au.v;
    #pragma unroll
    for (int tc = 0; tc < 8; ++tc) {
      const int nb = tc * 16;
      // B fragment (32x16 f16): lane holds col nb+lrow, K = half*16 + (0..15)
      const v16h bfrag = *(const v16h*)(&dYh[nb + lrow][half * 16]);
      v8f c = {};
      c = __builtin_amdgcn_wmma_f32_16x16x32_f16(false, afrag, false, bfrag,
                                                 (short)0, c, false, false);
      #pragma unroll
      for (int r = 0; r < 8; ++r) {
        inc[(mb + r + half * 8) * STRIDE + (nb + lrow)] = c[r];
      }
    }
  }

  // ---- Phase 3: anti-diagonal wavefront PDE; one thread per diagonal element ----
  diag[0][tid + 1] = 1.0f;   // prev2
  diag[1][tid + 1] = 1.0f;   // prev1
  if (tid == 0) { diag[0][0] = 0.0f; diag[1][0] = 0.0f; diag[2][0] = 0.0f; }
  __syncthreads();

  int p2 = 0, p1 = 1, cu = 2;
  for (int d = 2; d <= 2 * MMDIM; ++d) {
    int r  = tid - 1;      r  = min(max(r, 0),  MMDIM - 1);
    int cc = d - tid - 1;  cc = min(max(cc, 0), MMDIM - 1);
    float incr = inc[r * STRIDE + cc];
    float k10 = diag[p1][tid + 1];
    float k01 = diag[p1][tid];      // shift: element 0 is the fixed zero pad
    float k00 = diag[p2][tid];
    float i2  = incr * incr * (1.0f / 12.0f);
    float raw = (k10 + k01) * (1.0f + 0.5f * incr + i2) - k00 * (1.0f - i2);
    bool boundary = ((tid == 0) || (tid == d)) && (d <= MMDIM);
    diag[cu][tid + 1] = boundary ? 1.0f : raw;
    __syncthreads();
    int tmp = p2; p2 = p1; p1 = cu; cu = tmp;
  }
  if (tid == MMDIM) Kout[blk] = diag[p1][MMDIM + 1];
}

// Deterministic weighted reduction: mean(KXX) + mean(KYY) - 2*mean(KXY)
__global__ __launch_bounds__(256)
void sig_reduce_kernel(const float* __restrict__ K, float* __restrict__ out) {
  __shared__ float s[256];
  int tid = threadIdx.x;
  float acc = 0.0f;
  for (int i = tid; i < 3072; i += 256) {
    float w = (i < 2048) ? 1.0f : -2.0f;
    acc += w * K[i];
  }
  s[tid] = acc;
  __syncthreads();
  for (int off = 128; off > 0; off >>= 1) {
    if (tid < off) s[tid] += s[tid + off];
    __syncthreads();
  }
  if (tid == 0) out[0] = s[0] * (1.0f / 1024.0f);
}

extern "C" void kernel_launch(void* const* d_in, const int* in_sizes, int n_in,
                              void* d_out, int out_size, void* d_ws, size_t ws_size,
                              hipStream_t stream) {
  const float* X = (const float*)d_in[0];
  const float* Y = (const float*)d_in[1];
  float* Kbuf = (float*)d_ws;   // 3072 floats of per-pair kernel values
  sig_pde_fused_kernel<<<dim3(3072), dim3(128), 0, stream>>>(X, Y, Kbuf);
  sig_reduce_kernel<<<dim3(1), dim3(256), 0, stream>>>(Kbuf, (float*)d_out);
}
